// FaPN_17643725652298
// MI455X (gfx1250) — matrix operand order repaired and checked by reading, not compiled, not run on hardware
//
#include <hip/hip_runtime.h>
#include <math.h>

// FaPN on gfx1250 (MI455X). All matrix-shaped contractions (1x1 convs and the
// deformable-conv tap contraction) run on v_wmma_f32_16x16x32_bf16 with f32
// accumulation. Weights are pre-packed once per launch into WMMA-fragment-
// major bf16 (A operand = two coalesced global_load_b128, no LDS); only the
// B side (NCHW transpose / deformable gather) stages through LDS, fragment-
// major, so each B operand is two ds_load_b128. Workspace: ~134 MB.

#define EPS_GN 1e-5f

typedef __bf16  bf16x16 __attribute__((ext_vector_type(16)));
typedef float   f32x8   __attribute__((ext_vector_type(8)));

// Pack two f32 -> two bf16 (RNE) in one u32, pure 32-bit ALU ops.
__device__ inline unsigned int pack2bf(float a, float b) {
  union { float f; unsigned int u; } ca, cb; ca.f = a; cb.f = b;
  const unsigned int ra = ca.u + 0x7fffu + ((ca.u >> 16) & 1u);
  const unsigned int rb = cb.u + 0x7fffu + ((cb.u >> 16) & 1u);
  return (ra >> 16) | (rb & 0xffff0000u);
}

// Load one 16-element bf16 fragment stored fragment-major in LDS:
// lane's data = 32 contiguous bytes -> two ds_load_b128.
__device__ inline bf16x16 load_frag(const uint4* tile /* [32][2] quads */) {
  const int lane = threadIdx.x & 31;
  union { uint4 q[2]; bf16x16 b; } c;
  c.q[0] = tile[lane * 2];
  c.q[1] = tile[lane * 2 + 1];
  return c.b;
}

__device__ inline f32x8 wmma_bf16(bf16x16 a, bf16x16 b, f32x8 c) {
  return __builtin_amdgcn_wmma_f32_16x16x32_bf16(false, a, false, b,
                                                 (short)0, c, false, false);
}

// Fragment-index map (K order this code uses for both A and B; contiguous, not AMD's table order):
// A (16Mx32K): lane = hi*16 + r holds row M=r; half h of its 16 elements is
//              K = hi*16 + h*8 + (0..7)  -> 8 *consecutive* K values.
// B (32Kx16N): lane = hi*16 + n holds col N=n; half h is
//              K = hi*16 + h*8 + (0..7)  -> 8 consecutive K values.

// --------- one-time weight pack: [O,C] f32 -> fragment-major bf16 (uint4).
// dst unit u = ((om*KC + kc)*32 + lane)*2 + half ; 8 bf16 per unit.
__global__ __launch_bounds__(256) void k_pack_w(uint4* __restrict__ dst,
                                                const float* __restrict__ W,
                                                int KC, int C, int total) {
  const int u = blockIdx.x * 256 + threadIdx.x;
  if (u >= total) return;
  const int half = u & 1, lane = (u >> 1) & 31;
  const int t = u >> 6;
  const int kc = t % KC, om = t / KC;
  const int hi = lane >> 4, r = lane & 15;
  const float* src = &W[(om * 16 + r) * C + kc * 32 + hi * 16 + half * 8];
  const float4 w0 = ((const float4*)src)[0];
  const float4 w1 = ((const float4*)src)[1];
  uint4 q;
  q.x = pack2bf(w0.x, w0.y); q.y = pack2bf(w0.z, w0.w);
  q.z = pack2bf(w1.x, w1.y); q.w = pack2bf(w1.z, w1.w);
  dst[u] = q;
}

// ---------------------------------------------------------------- avg pool
__global__ __launch_bounds__(256) void k_avgpool(const float* __restrict__ x,
                                                 float* __restrict__ pooled,
                                                 int HW) {
  __shared__ float s[256];
  const float* p = x + (size_t)blockIdx.x * HW;
  float acc = 0.f;
  for (int i = threadIdx.x; i < HW; i += 256) acc += p[i];
  s[threadIdx.x] = acc; __syncthreads();
  for (int st = 128; st > 0; st >>= 1) {
    if (threadIdx.x < st) s[threadIdx.x] += s[threadIdx.x + st];
    __syncthreads();
  }
  if (threadIdx.x == 0) pooled[blockIdx.x] = s[0] / (float)HW;
}

// ------------------------------------- attention: 1x1 conv + GN8 + sigmoid
__global__ __launch_bounds__(64) void k_atten(const float* __restrict__ pooled,
                                              const float* __restrict__ aw,
                                              const float* __restrict__ ab,
                                              const float* __restrict__ g1w,
                                              const float* __restrict__ g1b,
                                              float* __restrict__ atten, int C) {
  const int Cg = C >> 3;
  const int b = blockIdx.x >> 3, g = blockIdx.x & 7;
  __shared__ float sy[64], s1[64], s2[64];
  const int j = threadIdx.x;
  float y = 0.f;
  if (j < Cg) {
    const int c = g * Cg + j;
    float acc = ab[c];
    for (int ci = 0; ci < C; ++ci) acc += aw[c * C + ci] * pooled[b * C + ci];
    y = acc;
  }
  sy[j] = y; s1[j] = (j < Cg) ? y : 0.f; s2[j] = (j < Cg) ? y * y : 0.f;
  __syncthreads();
  for (int st = 32; st > 0; st >>= 1) {
    if (j < st) { s1[j] += s1[j + st]; s2[j] += s2[j + st]; }
    __syncthreads();
  }
  const float mu  = s1[0] / (float)Cg;
  const float var = s2[0] / (float)Cg - mu * mu;
  const float inv = rsqrtf(var + EPS_GN);
  if (j < Cg) {
    const int c = g * Cg + j;
    const float xn = (sy[j] - mu) * inv * g1w[c] + g1b[c];
    atten[b * C + c] = 1.f / (1.f + expf(-xn));
  }
}

// ----------------------- generic 1x1-conv GEMM: out[b,o,hw] = W[o,:]@X[b,:,hw]
// Wp = fragment-major bf16 packed weights. Optional per-(b,c) scale (1+atten).
// Requires: HW % 64 == 0, O % 64 == 0, C % 32 == 0.
__global__ __launch_bounds__(256) void k_gemm_bf16(
    float* __restrict__ out, const float* __restrict__ X,
    const uint4* __restrict__ Wp, const float* __restrict__ bias,
    const float* __restrict__ scale, int C, int O, int HW) {
  __shared__ uint4 sB[4 * 32 * 2];   // B: 32 K x 64 pixels, fragment-major
  const int tid = threadIdx.x;
  const int lane = tid & 31;
  const int waveId = tid >> 5;
  const int wave_m = waveId & 3, wave_n = waveId >> 2;   // 4x2 wave grid
  const int ntile = blockIdx.x * 64, o0 = blockIdx.y * 64;
  const int bImg = ntile / HW, hw0 = ntile - bImg * HW;
  const int KC = C >> 5;
  const uint4* wp = Wp + (size_t)(((o0 >> 4) + wave_m) * KC) * 64 + lane * 2;

  // B fill-unit decomposition (256 units == 256 threads)
  const int u_sub  = tid >> 6;
  const int u_lane = (tid >> 1) & 31;
  const int u_half = tid & 1;
  const int n      = u_sub * 16 + (u_lane & 15);
  const int c0base = (u_lane >> 4) * 16 + u_half * 8;

  f32x8 acc0, acc1;
#pragma unroll
  for (int i = 0; i < 8; ++i) { acc0[i] = 0.f; acc1[i] = 0.f; }

  for (int kk = 0; kk < C; kk += 32) {
    // A fragment: two coalesced b128 loads from packed weights (L2-resident).
    union { uint4 q[2]; bf16x16 b; } af;
    af.q[0] = wp[0]; af.q[1] = wp[1];
    // B: col n, 8 consecutive channels (stride-HW walk).
    {
      const int c0 = kk + c0base;
      const float* p = &X[(bImg * C + c0) * HW + hw0 + n];
      float v[8];
#pragma unroll
      for (int j = 0; j < 8; ++j) { v[j] = *p; p += HW; }
      if (scale) {
        const float4* sp = (const float4*)&scale[bImg * C + c0];
        const float4 s0 = sp[0], s1 = sp[1];
        v[0] *= 1.f + s0.x; v[1] *= 1.f + s0.y;
        v[2] *= 1.f + s0.z; v[3] *= 1.f + s0.w;
        v[4] *= 1.f + s1.x; v[5] *= 1.f + s1.y;
        v[6] *= 1.f + s1.z; v[7] *= 1.f + s1.w;
      }
      uint4 q;
      q.x = pack2bf(v[0], v[1]); q.y = pack2bf(v[2], v[3]);
      q.z = pack2bf(v[4], v[5]); q.w = pack2bf(v[6], v[7]);
      sB[tid] = q;
    }
    if (kk + 32 < C) __builtin_prefetch(wp + 64, 0, 1);  // global_prefetch_b8
    __syncthreads();
    const bf16x16 b0 = load_frag(&sB[(wave_n * 2) * 64]);
    const bf16x16 b1 = load_frag(&sB[(wave_n * 2 + 1) * 64]);
    acc0 = wmma_bf16(af.b, b0, acc0);
    acc1 = wmma_bf16(af.b, b1, acc1);
    __syncthreads();
    wp += 64;
  }

  const int hi = lane >> 4, nc = lane & 15;
#pragma unroll
  for (int r = 0; r < 8; ++r) {               // C layout: M = r + 8*hi, N = nc
    const int o = o0 + wave_m * 16 + r + hi * 8;
    const float bv = bias ? bias[o] : 0.f;
    const int base = (bImg * O + o) * HW + hw0 + wave_n * 32 + nc;
    out[base]      = acc0[r] + bv;
    out[base + 16] = acc1[r] + bv;
  }
}

// ------------------------------------------------------- GroupNorm two-pass
__global__ __launch_bounds__(256) void k_gn_stats(const float* __restrict__ x,
                                                  float* __restrict__ stats,
                                                  int C, int G, int HW) {
  const int Cg = C / G;
  const int b = blockIdx.x / G, g = blockIdx.x - b * G;
  const size_t base = ((size_t)b * C + (size_t)g * Cg) * HW;
  const int n = Cg * HW;
  __shared__ float s1[256], s2[256];
  float a1 = 0.f, a2 = 0.f;
  for (int i = threadIdx.x; i < n; i += 256) {
    const float v = x[base + i]; a1 += v; a2 += v * v;
  }
  s1[threadIdx.x] = a1; s2[threadIdx.x] = a2; __syncthreads();
  for (int st = 128; st > 0; st >>= 1) {
    if (threadIdx.x < st) { s1[threadIdx.x] += s1[threadIdx.x + st];
                            s2[threadIdx.x] += s2[threadIdx.x + st]; }
    __syncthreads();
  }
  if (threadIdx.x == 0) {
    const float mu = s1[0] / (float)n;
    const float var = s2[0] / (float)n - mu * mu;
    stats[blockIdx.x * 2]     = mu;
    stats[blockIdx.x * 2 + 1] = rsqrtf(var + EPS_GN);
  }
}

__global__ void k_gn_apply(float* __restrict__ dst, const float* __restrict__ src,
                           const float* __restrict__ stats,
                           const float* __restrict__ gw, const float* __restrict__ gb,
                           int C, int G, int HW, int relu, int total) {
  const int i = blockIdx.x * blockDim.x + threadIdx.x;
  if (i >= total) return;
  const int c = (i / HW) % C;
  const int b = i / (HW * C);
  const int g = c / (C / G);
  const float mu = stats[(b * G + g) * 2], inv = stats[(b * G + g) * 2 + 1];
  float v = (src[i] - mu) * inv * gw[c] + gb[c];
  if (relu) v = fmaxf(v, 0.f);
  dst[i] = v;
}

// ---------------------------------- bilinear 2x upsample, half-pixel centers
__global__ void k_upsample2x(float* __restrict__ out, const float* __restrict__ in,
                             int Hin, int Win, int total) {
  const int i = blockIdx.x * blockDim.x + threadIdx.x;
  if (i >= total) return;
  const int Wout = 2 * Win, Hout = 2 * Hin;
  const int wo = i % Wout;
  const int ho = (i / Wout) % Hout;
  const int bc = i / (Wout * Hout);
  const float sy = (ho + 0.5f) * 0.5f - 0.5f;
  const float sx = (wo + 0.5f) * 0.5f - 0.5f;
  const float y0 = floorf(sy), x0 = floorf(sx);
  const float fy = sy - y0, fx = sx - x0;
  const int y0i = max(0, min(Hin - 1, (int)y0));
  const int y1i = max(0, min(Hin - 1, (int)y0 + 1));
  const int x0i = max(0, min(Win - 1, (int)x0));
  const int x1i = max(0, min(Win - 1, (int)x0 + 1));
  const float* p = in + (size_t)bc * Hin * Win;
  const float v =
      (1.f - fy) * ((1.f - fx) * p[y0i * Win + x0i] + fx * p[y0i * Win + x1i]) +
      fy        * ((1.f - fx) * p[y1i * Win + x0i] + fx * p[y1i * Win + x1i]);
  out[i] = v;
}

// ------------------- offset conv: 18ch 1x1 over concat([arm, 2*up]) (tiny O)
__global__ void k_offconv(float* __restrict__ out, const float* __restrict__ arm,
                          const float* __restrict__ up, const float* __restrict__ ow,
                          const float* __restrict__ ob,
                          int C1, int C2, int HW, int total) {
  const int i = blockIdx.x * blockDim.x + threadIdx.x;
  if (i >= total) return;
  const int hw = i % HW;
  const int j  = (i / HW) % 18;
  const int b  = i / (18 * HW);
  const float* wr = ow + j * (C1 + C2);
  float acc = ob[j];
  const float* pa = arm + b * C1 * HW + hw;
  for (int c = 0; c < C1; ++c) { acc += wr[c] * (*pa); pa += HW; }
  const float* pu = up + b * C2 * HW + hw;
  for (int c = 0; c < C2; ++c) { acc += wr[C1 + c] * 2.f * (*pu); pu += HW; }
  out[i] = acc;
}

// -------- deformable 3x3 conv as WMMA GEMM over K = C*9 with gathered B-tile
// out[b,o,hw] = relu( sum_{c,k} W[o,c*9+k] * bilinear(feat[b,c], tap k) + db[o] )
//               + arm[b,o,hw]
// Wp = fragment-major bf16 packed [O, C*9] weights.
// Requires HW % 32 == 0, O % 64 == 0, (C*9) % 32 == 0.
__global__ __launch_bounds__(128) void k_dcn_bf16(
    float* __restrict__ out, const float* __restrict__ feat,
    const float* __restrict__ off, const uint4* __restrict__ Wp,
    const float* __restrict__ bias, const float* __restrict__ arm,
    int C, int O, int H, int W) {
  const int HW = H * W;
  const int CK = C * 9;
  __shared__ int   sIdx[32 * 9 * 4];
  __shared__ float sWgt[32 * 9 * 4];
  __shared__ uint4 sB[2 * 32 * 2];   // 32 K x 32 pixels, fragment-major
  const int tid = threadIdx.x;
  const int lane = tid & 31;
  const int waveId = tid >> 5;                 // 4 waves: O sub-tile of 16
  const int n0 = blockIdx.x * 32;
  const int bImg = n0 / HW, hw0 = n0 - bImg * HW;
  const int o0 = blockIdx.y * 64;
  const int KC = CK >> 5;
  const uint4* wp = Wp + (size_t)(((o0 >> 4) + waveId) * KC) * 64 + lane * 2;

  // Per-tile sampling metadata: 32 pixels x 9 taps x 4 bilinear corners.
  for (int e = tid; e < 32 * 9; e += 128) {
    const int pix = e / 9, tap = e - pix * 9;
    const int hw = hw0 + pix;
    const int h = hw / W, w = hw - h * W;
    const float dy = off[(bImg * 18 + 2 * tap)     * HW + hw];
    const float dx = off[(bImg * 18 + 2 * tap + 1) * HW + hw];
    const float py = (float)(h + tap / 3 - 1) + dy;
    const float px = (float)(w + tap % 3 - 1) + dx;
    const float y0 = floorf(py), x0 = floorf(px);
#pragma unroll
    for (int j = 0; j < 4; ++j) {
      const float yc = y0 + (float)(j >> 1);
      const float xc = x0 + (float)(j & 1);
      float wgt = (1.f - fabsf(py - yc)) * (1.f - fabsf(px - xc));
      const bool valid = (yc >= 0.f) && (yc <= (float)(H - 1)) &&
                         (xc >= 0.f) && (xc <= (float)(W - 1));
      if (!valid) wgt = 0.f;                   // zeros outside
      const int yi = (int)fminf(fmaxf(yc, 0.f), (float)(H - 1));
      const int xi = (int)fminf(fmaxf(xc, 0.f), (float)(W - 1));
      sIdx[e * 4 + j] = yi * W + xi;
      sWgt[e * 4 + j] = wgt;
    }
  }
  __syncthreads();

  const int u_lane = (tid >> 1) & 31;
  const int u_half = tid & 1;
  const int pixF   = (tid >> 6) * 16 + (u_lane & 15);
  const int ckBase = (u_lane >> 4) * 16 + u_half * 8;

  f32x8 acc0, acc1;
#pragma unroll
  for (int i = 0; i < 8; ++i) { acc0[i] = 0.f; acc1[i] = 0.f; }

  for (int kk = 0; kk < CK; kk += 32) {
    // A fragment: two coalesced b128 loads from packed weights.
    union { uint4 q[2]; bf16x16 b; } af;
    af.q[0] = wp[0]; af.q[1] = wp[1];
    // B: 8 consecutive ck values at one pixel (4-corner gather each).
    {
      const int ck0 = kk + ckBase;
      float v[8];
#pragma unroll
      for (int j = 0; j < 8; ++j) {
        const int ck = ck0 + j;
        const int c = ck / 9, tap = ck - c * 9;
        const float* fb = feat + (bImg * C + c) * HW;
        const int m = (pixF * 9 + tap) * 4;
        v[j] = sWgt[m]     * fb[sIdx[m]]     + sWgt[m + 1] * fb[sIdx[m + 1]] +
               sWgt[m + 2] * fb[sIdx[m + 2]] + sWgt[m + 3] * fb[sIdx[m + 3]];
      }
      uint4 q;
      q.x = pack2bf(v[0], v[1]); q.y = pack2bf(v[2], v[3]);
      q.z = pack2bf(v[4], v[5]); q.w = pack2bf(v[6], v[7]);
      sB[tid] = q;
    }
    if (kk + 32 < CK) __builtin_prefetch(wp + 64, 0, 1);  // global_prefetch_b8
    __syncthreads();
    const bf16x16 b0 = load_frag(&sB[0]);
    const bf16x16 b1 = load_frag(&sB[64]);
    acc0 = wmma_bf16(af.b, b0, acc0);
    acc1 = wmma_bf16(af.b, b1, acc1);
    __syncthreads();
    wp += 64;
  }

  const int hi = lane >> 4, nc = lane & 15;
#pragma unroll
  for (int r = 0; r < 8; ++r) {
    const int o = o0 + waveId * 16 + r + hi * 8;
    const int base = (bImg * O + o) * HW + hw0;
    out[base + nc]      = fmaxf(acc0[r] + bias[o], 0.f) + arm[base + nc];
    out[base + 16 + nc] = fmaxf(acc1[r] + bias[o], 0.f) + arm[base + 16 + nc];
  }
}

// ===========================================================================
extern "C" void kernel_launch(void* const* d_in, const int* in_sizes, int n_in,
                              void* d_out, int out_size, void* d_ws, size_t ws_size,
                              hipStream_t stream) {
  (void)in_sizes; (void)n_in; (void)out_size; (void)ws_size;
  const float* src0 = (const float*)d_in[0];   // [4,256,64,64]
  const float* src1 = (const float*)d_in[1];   // [4,512,32,32]
  const float* src2 = (const float*)d_in[2];   // [4,1024,16,16]
  const float* const* in = (const float* const*)d_in;
  float* ws  = (float*)d_ws;
  float* outP = (float*)d_out;

  // workspace layout (floats; every offset is a multiple of 4 -> 16B aligned)
  const size_t oPooled = 0;                    // 2048
  const size_t oAtten  = oPooled + 2048;       // 2048
  const size_t oStats  = oAtten  + 2048;       // 128
  const size_t oTmp    = oStats  + 128;        // 4,194,304 (shared pre-GN buf)
  const size_t oArm0   = oTmp    + 4194304;    // 2,097,152
  const size_t oUp0    = oArm0   + 2097152;    // 4,194,304
  const size_t oOffT   = oUp0    + 4194304;    // 294,912
  const size_t oOff    = oOffT   + 294912;     // 294,912
  const size_t oA0     = oOff    + 294912;     // 2,097,152
  const size_t oArm1   = oA0     + 2097152;    // 4,194,304
  const size_t oUp1    = oArm1   + 4194304;    // 8,388,608
  const size_t oA1     = oUp1    + 8388608;    // 4,194,304
  // packed bf16 weights (u32 counts)
  const size_t oPkF0   = oA1     + 4194304;    // 512x512   -> 131,072
  const size_t oPkF1   = oPkF0   + 131072;     // 256x256   -> 32,768
  const size_t oPkD0   = oPkF1   + 32768;      // 512x9216  -> 2,359,296
  const size_t oPkD1   = oPkD0   + 2359296;    // 256x4608  -> 589,824
  const size_t oPkB0   = oPkD1   + 589824;     // 256x1024  -> 131,072
  const size_t oPkB1   = oPkB0   + 131072;     // 256x512   -> 65,536
  const size_t oPkB2   = oPkB1   + 65536;      // 256x256   -> 32,768

  // ---------------- one-time weight packing (fragment-major bf16)
  {
    struct PW { size_t off; int idx, O, C; };
    const PW pw[7] = { {oPkF0, 7, 512, 512},  {oPkF1, 21, 256, 256},
                       {oPkD0, 15, 512, 9216},{oPkD1, 29, 256, 4608},
                       {oPkB0, 31, 256, 1024},{oPkB1, 35, 256, 512},
                       {oPkB2, 39, 256, 256} };
    for (int t = 0; t < 7; ++t) {
      const int KC = pw[t].C >> 5;
      const int total = (pw[t].O >> 4) * KC * 64;
      k_pack_w<<<(total + 255) / 256, 256, 0, stream>>>(
          (uint4*)(ws + pw[t].off), in[pw[t].idx], KC, pw[t].C, total);
    }
  }

  // ---------------- Align stage 0: feat_l = src1 (512@32x32), feat_s = src2
  k_avgpool<<<4 * 512, 256, 0, stream>>>(src1, ws + oPooled, 1024);
  k_atten<<<32, 64, 0, stream>>>(ws + oPooled, in[3], in[4], in[5], in[6],
                                 ws + oAtten, 512);
  k_gemm_bf16<<<dim3(4096 / 64, 512 / 64), 256, 0, stream>>>(
      ws + oTmp, src1, (const uint4*)(ws + oPkF0), in[8], ws + oAtten,
      512, 512, 1024);
  k_gn_stats<<<4 * 8, 256, 0, stream>>>(ws + oTmp, ws + oStats, 512, 8, 1024);
  k_gn_apply<<<(4 * 512 * 1024 + 255) / 256, 256, 0, stream>>>(
      ws + oArm0, ws + oTmp, ws + oStats, in[9], in[10], 512, 8, 1024, 0,
      4 * 512 * 1024);
  k_upsample2x<<<(4 * 1024 * 1024 + 255) / 256, 256, 0, stream>>>(
      ws + oUp0, src2, 16, 16, 4 * 1024 * 1024);
  k_offconv<<<(4 * 18 * 1024 + 255) / 256, 256, 0, stream>>>(
      ws + oOffT, ws + oArm0, ws + oUp0, in[11], in[12], 512, 1024, 1024,
      4 * 18 * 1024);
  k_gn_stats<<<4 * 9, 256, 0, stream>>>(ws + oOffT, ws + oStats, 18, 9, 1024);
  k_gn_apply<<<(4 * 18 * 1024 + 255) / 256, 256, 0, stream>>>(
      ws + oOff, ws + oOffT, ws + oStats, in[13], in[14], 18, 9, 1024, 0,
      4 * 18 * 1024);
  k_dcn_bf16<<<dim3(4 * 1024 / 32, 512 / 64), 128, 0, stream>>>(
      ws + oA0, ws + oUp0, ws + oOff, (const uint4*)(ws + oPkD0), in[16],
      ws + oArm0, 1024, 512, 32, 32);

  // ---------------- Align stage 1: feat_l = src0 (256@64x64), feat_s = a0
  k_avgpool<<<4 * 256, 256, 0, stream>>>(src0, ws + oPooled, 4096);
  k_atten<<<32, 64, 0, stream>>>(ws + oPooled, in[17], in[18], in[19], in[20],
                                 ws + oAtten, 256);
  k_gemm_bf16<<<dim3(16384 / 64, 256 / 64), 256, 0, stream>>>(
      ws + oTmp, src0, (const uint4*)(ws + oPkF1), in[22], ws + oAtten,
      256, 256, 4096);
  k_gn_stats<<<4 * 8, 256, 0, stream>>>(ws + oTmp, ws + oStats, 256, 8, 4096);
  k_gn_apply<<<(4 * 256 * 4096 + 255) / 256, 256, 0, stream>>>(
      ws + oArm1, ws + oTmp, ws + oStats, in[23], in[24], 256, 8, 4096, 0,
      4 * 256 * 4096);
  k_upsample2x<<<(4 * 512 * 4096 + 255) / 256, 256, 0, stream>>>(
      ws + oUp1, ws + oA0, 32, 32, 4 * 512 * 4096);
  k_offconv<<<(4 * 18 * 4096 + 255) / 256, 256, 0, stream>>>(
      ws + oOffT, ws + oArm1, ws + oUp1, in[25], in[26], 256, 512, 4096,
      4 * 18 * 4096);
  k_gn_stats<<<4 * 9, 256, 0, stream>>>(ws + oOffT, ws + oStats, 18, 9, 4096);
  k_gn_apply<<<(4 * 18 * 4096 + 255) / 256, 256, 0, stream>>>(
      ws + oOff, ws + oOffT, ws + oStats, in[27], in[28], 18, 9, 4096, 0,
      4 * 18 * 4096);
  k_dcn_bf16<<<dim3(4 * 4096 / 32, 256 / 64), 128, 0, stream>>>(
      ws + oA1, ws + oUp1, ws + oOff, (const uint4*)(ws + oPkD1), in[30],
      ws + oArm1, 512, 256, 64, 64);

  // ---------------- Bottom-up heads: 1x1 conv + GN8 + ReLU -> d_out
  k_gemm_bf16<<<dim3(1024 / 64, 256 / 64), 256, 0, stream>>>(
      ws + oTmp, src2, (const uint4*)(ws + oPkB0), in[32], nullptr,
      1024, 256, 256);
  k_gn_stats<<<4 * 8, 256, 0, stream>>>(ws + oTmp, ws + oStats, 256, 8, 256);
  k_gn_apply<<<(262144 + 255) / 256, 256, 0, stream>>>(
      outP, ws + oTmp, ws + oStats, in[33], in[34], 256, 8, 256, 1, 262144);

  k_gemm_bf16<<<dim3(4096 / 64, 256 / 64), 256, 0, stream>>>(
      ws + oTmp, ws + oA0, (const uint4*)(ws + oPkB1), in[36], nullptr,
      512, 256, 1024);
  k_gn_stats<<<4 * 8, 256, 0, stream>>>(ws + oTmp, ws + oStats, 256, 8, 1024);
  k_gn_apply<<<(1048576 + 255) / 256, 256, 0, stream>>>(
      outP + 262144, ws + oTmp, ws + oStats, in[37], in[38], 256, 8, 1024, 1,
      1048576);

  k_gemm_bf16<<<dim3(16384 / 64, 256 / 64), 256, 0, stream>>>(
      ws + oTmp, ws + oA1, (const uint4*)(ws + oPkB2), in[40], nullptr,
      256, 256, 4096);
  k_gn_stats<<<4 * 8, 256, 0, stream>>>(ws + oTmp, ws + oStats, 256, 8, 4096);
  k_gn_apply<<<(4194304 + 255) / 256, 256, 0, stream>>>(
      outP + 1310720, ws + oTmp, ws + oStats, in[41], in[42], 256, 8, 4096, 1,
      4194304);
}
